// SMILESVAE_78005196030501
// MI455X (gfx1250) — matrix operand order repaired and checked by reading, not compiled, not run on hardware
//
#include <hip/hip_runtime.h>

// SMILES VAE on gfx1250: f16 WMMA (v_wmma_f32_16x16x32_f16) for every GEMM,
// fused LSTM cell (i/f/g/o tiles element-aligned per lane -> lane-local VALU
// nonlinearity), f32 cell state. ~1200 small launches amortized by graph replay.

typedef __attribute__((ext_vector_type(16))) _Float16 v16h;
typedef __attribute__((ext_vector_type(8)))  _Float16 v8h;
typedef __attribute__((ext_vector_type(8)))  float    v8f;

#define V_ 128
#define H_ 512
#define L_ 128
#define B_ 256
#define T_ 128

// ---------------------------------------------------------------------------
// Fragment loaders matching CDNA5 WMMA VGPR layouts (wave32).
// A: 16x32 f16, lane m=lane&15 holds row m; chunks at k+kh*8 and k+16+kh*8.
// B: 32x16 f16 (= W^T tile), lane n=lane&15 holds weight row n0+n,
//    chunks at k0+kh*8 and k0+16+kh*8 (same k order as A).
// ---------------------------------------------------------------------------
__device__ __forceinline__ v16h load_a_frag(const _Float16* __restrict__ base,
                                            int pitch, int row0, int k0, int lane) {
  int m = lane & 15, kh = lane >> 4;
  const _Float16* p = base + (size_t)(row0 + m) * pitch + k0;
  v8h c0 = *reinterpret_cast<const v8h*>(p + kh * 8);
  v8h c1 = *reinterpret_cast<const v8h*>(p + 16 + kh * 8);
  return __builtin_shufflevector(c0, c1, 0,1,2,3,4,5,6,7,8,9,10,11,12,13,14,15);
}

__device__ __forceinline__ v16h load_b_frag(const _Float16* __restrict__ wbase,
                                            int pitch, int n0, int k0, int lane) {
  int n = lane & 15, kh = lane >> 4;
  const _Float16* p = wbase + (size_t)(n0 + n) * pitch + k0;
  v8h c0 = *reinterpret_cast<const v8h*>(p + kh * 8);
  v8h c1 = *reinterpret_cast<const v8h*>(p + 16 + kh * 8);
  return __builtin_shufflevector(c0, c1, 0,1,2,3,4,5,6,7,8,9,10,11,12,13,14,15);
}

__device__ __forceinline__ float sigf(float x) {
  return 1.0f / (1.0f + __expf(-x));
}

// ---------------------------------------------------------------------------
// Fused LSTM cell step: g = [x_t, h_prev] @ Wcat^T + b ; apply gates.
// Block = 128 threads = 4 waves; block tile = 32 rows x 32 h-cols.
// Wave w computes the SAME 16x16 (b,h) patch in all 4 gate blocks, so the
// four v8f accumulators are element-aligned -> lane-local cell update.
// Grid: (B/32, H/32) = (8, 16).
// ---------------------------------------------------------------------------
__global__ __launch_bounds__(128)
void lstm_cell_wmma(const _Float16* __restrict__ x, int xpitch, int Kx,
                    const _Float16* __restrict__ hprev,
                    const _Float16* __restrict__ W, int wpitch,
                    const float* __restrict__ bias,
                    float* __restrict__ cbuf,
                    _Float16* __restrict__ hout,
                    _Float16* __restrict__ yout, int ypitch) {
  const int wave = threadIdx.x >> 5, lane = threadIdx.x & 31;
  const int row0  = blockIdx.x * 32 + (wave >> 1) * 16;   // batch tile
  const int hcol0 = blockIdx.y * 32 + (wave & 1) * 16;    // hidden tile

  v8f acc[4] = {};

  // Prefetch the first weight rows of the i and o gate tiles (global_prefetch_b8).
  __builtin_prefetch(W + (size_t)(hcol0 + (lane & 15)) * wpitch, 0, 3);
  __builtin_prefetch(W + (size_t)(3 * H_ + hcol0 + (lane & 15)) * wpitch, 0, 3);

  // x contribution (weight cols [0, Kx))
  for (int k0 = 0; k0 < Kx; k0 += 32) {
    v16h a = load_a_frag(x, xpitch, row0, k0, lane);
#pragma unroll
    for (int g = 0; g < 4; ++g) {
      v16h b = load_b_frag(W, wpitch, g * H_ + hcol0, k0, lane);
      acc[g] = __builtin_amdgcn_wmma_f32_16x16x32_f16(
          false, a, false, b, (short)0, acc[g], false, false);
    }
  }
  // h contribution (weight cols [Kx, Kx+H))
  for (int k0 = 0; k0 < H_; k0 += 32) {
    v16h a = load_a_frag(hprev, H_, row0, k0, lane);
#pragma unroll
    for (int g = 0; g < 4; ++g) {
      v16h b = load_b_frag(W, wpitch, g * H_ + hcol0, Kx + k0, lane);
      acc[g] = __builtin_amdgcn_wmma_f32_16x16x32_f16(
          false, a, false, b, (short)0, acc[g], false, false);
    }
  }

  // Cell update: C/D layout -> lane holds col n=lane&15, rows r+(lane>>4)*8.
  const int n = lane & 15, kh = lane >> 4;
  const int hg = hcol0 + n;
  const float bi = bias[hg];
  const float bf = bias[H_ + hg];
  const float bg = bias[2 * H_ + hg];
  const float bo = bias[3 * H_ + hg];
#pragma unroll
  for (int r = 0; r < 8; ++r) {
    const int brow = row0 + kh * 8 + r;
    const size_t idx = (size_t)brow * H_ + hg;
    float gi = acc[0][r] + bi;
    float gf = acc[1][r] + bf;
    float gg = acc[2][r] + bg;
    float go = acc[3][r] + bo;
    float cn = sigf(gf) * cbuf[idx] + sigf(gi) * tanhf(gg);
    float hn = sigf(go) * tanhf(cn);
    cbuf[idx] = cn;
    hout[idx] = (_Float16)hn;
    if (yout) yout[(size_t)brow * ypitch + hg] = (_Float16)hn;
  }
}

// ---------------------------------------------------------------------------
// Generic WMMA GEMM: C[M,N] = A[M,K](f16) @ W[N,K](f16)^T + bias(f32).
// Optional f32 and/or f16 outputs with independent pitches.
// Block 128 thr, tile 32x32, grid (M/32, N/32).
// ---------------------------------------------------------------------------
__global__ __launch_bounds__(128)
void gemm_wmma(const _Float16* __restrict__ A, int apitch,
               const _Float16* __restrict__ W, int wpitch, int K,
               const float* __restrict__ bias,
               float* __restrict__ outf, int opitchf,
               _Float16* __restrict__ outh, int opitchh) {
  const int wave = threadIdx.x >> 5, lane = threadIdx.x & 31;
  const int row0 = blockIdx.x * 32 + (wave >> 1) * 16;
  const int n0   = blockIdx.y * 32 + (wave & 1) * 16;

  v8f acc = {};
  __builtin_prefetch(W + (size_t)(n0 + (lane & 15)) * wpitch, 0, 3);
  for (int k0 = 0; k0 < K; k0 += 32) {
    v16h a = load_a_frag(A, apitch, row0, k0, lane);
    v16h b = load_b_frag(W, wpitch, n0, k0, lane);
    acc = __builtin_amdgcn_wmma_f32_16x16x32_f16(
        false, a, false, b, (short)0, acc, false, false);
  }
  const int n = lane & 15, kh = lane >> 4;
  const int ng = n0 + n;
  const float bv = bias ? bias[ng] : 0.0f;
#pragma unroll
  for (int r = 0; r < 8; ++r) {
    const int mg = row0 + kh * 8 + r;
    float v = acc[r] + bv;
    if (outf) outf[(size_t)mg * opitchf + ng] = v;
    if (outh) outh[(size_t)mg * opitchh + ng] = (_Float16)v;
  }
}

// ---------------------------------------------------------------------------
// Small helper kernels
// ---------------------------------------------------------------------------
__global__ void cvt2d_k(_Float16* __restrict__ dst, int dpitch,
                        const float* __restrict__ src, int rows, int cols) {
  int idx = blockIdx.x * blockDim.x + threadIdx.x;
  if (idx >= rows * cols) return;
  int r = idx / cols, c = idx - r * cols;
  dst[(size_t)r * dpitch + c] = (_Float16)src[idx];
}

// seq[(t*B + b)*H + h] = emb16[x[b*T + t]*H + h]
__global__ void embed_seq_k(_Float16* __restrict__ seq, const int* __restrict__ x,
                            const _Float16* __restrict__ emb16) {
  int idx = blockIdx.x * blockDim.x + threadIdx.x;
  if (idx >= T_ * B_ * H_) return;
  int h = idx & (H_ - 1);
  int tb = idx >> 9;            // H_ = 512
  int b = tb & (B_ - 1);
  int t = tb >> 8;              // B_ = 256
  int tok = x[b * T_ + t];
  seq[idx] = emb16[(size_t)tok * H_ + h];
}

// e[b*H + h] = emb16[tok[b]*H + h]
__global__ void embed_tok_k(_Float16* __restrict__ e, const int* __restrict__ tok,
                            const _Float16* __restrict__ emb16) {
  int idx = blockIdx.x * blockDim.x + threadIdx.x;
  if (idx >= B_ * H_) return;
  int h = idx & (H_ - 1);
  int b = idx >> 9;
  e[idx] = emb16[(size_t)tok[b] * H_ + h];
}

// First-max argmax over V_=128 logits per row (matches jnp.argmax tie-break).
__global__ void argmax_k(int* __restrict__ tok, const float* __restrict__ logits,
                         int rowpitch) {
  __shared__ float sv[128];
  __shared__ int   si[128];
  int b = blockIdx.x, v = threadIdx.x;
  sv[v] = logits[(size_t)b * rowpitch + v];
  si[v] = v;
  __syncthreads();
  for (int s = 64; s > 0; s >>= 1) {
    if (v < s) {
      float ov = sv[v + s]; int oi = si[v + s];
      if (ov > sv[v] || (ov == sv[v] && oi < si[v])) { sv[v] = ov; si[v] = oi; }
    }
    __syncthreads();
  }
  if (v == 0) tok[b] = si[0];
}

__global__ void fill_int_k(int* __restrict__ p, int val, int n) {
  int i = blockIdx.x * blockDim.x + threadIdx.x;
  if (i < n) p[i] = val;
}

// ---------------------------------------------------------------------------
// Host orchestration
// ---------------------------------------------------------------------------
extern "C" void kernel_launch(void* const* d_in, const int* in_sizes, int n_in,
                              void* d_out, int out_size, void* d_ws, size_t ws_size,
                              hipStream_t stream) {
  (void)in_sizes; (void)n_in; (void)out_size; (void)ws_size;

  const int*   x           = (const int*)d_in[0];
  const float* emb         = (const float*)d_in[1];
  const float* enc_wih_l0  = (const float*)d_in[2];
  const float* enc_whh_l0  = (const float*)d_in[3];
  const float* enc_b_l0    = (const float*)d_in[4];
  const float* enc_wih_l1  = (const float*)d_in[5];
  const float* enc_whh_l1  = (const float*)d_in[6];
  const float* enc_b_l1    = (const float*)d_in[7];
  const float* fc_mu_w     = (const float*)d_in[8];
  const float* fc_mu_b     = (const float*)d_in[9];
  const float* fc_lv_w     = (const float*)d_in[10];
  const float* fc_lv_b     = (const float*)d_in[11];
  const float* dec_in_w    = (const float*)d_in[12];
  const float* dec_in_b    = (const float*)d_in[13];
  const float* dec_wih     = (const float*)d_in[14];
  const float* dec_whh     = (const float*)d_in[15];
  const float* dec_b       = (const float*)d_in[16];
  const float* dec_out_w   = (const float*)d_in[17];
  const float* dec_out_b   = (const float*)d_in[18];

  float* out = (float*)d_out;

  // --- workspace carve-out (deterministic layout, ~130 MB) ---
  char* wsb = (char*)d_ws;
  size_t off = 0;
  auto alloc = [&](size_t bytes) -> void* {
    void* p = wsb + off;
    off = (off + bytes + 255) & ~(size_t)255;
    return p;
  };
  _Float16* emb16   = (_Float16*)alloc((size_t)V_ * H_ * 2);
  _Float16* seq16   = (_Float16*)alloc((size_t)T_ * B_ * H_ * 2);        // 32 MB
  _Float16* seq1    = (_Float16*)alloc((size_t)T_ * B_ * 2 * H_ * 2);    // 64 MB
  _Float16* Wenc0f  = (_Float16*)alloc((size_t)4 * H_ * 1024 * 2);
  _Float16* Wenc0b  = (_Float16*)alloc((size_t)4 * H_ * 1024 * 2);
  _Float16* Wenc1f  = (_Float16*)alloc((size_t)4 * H_ * 1536 * 2);
  _Float16* Wenc1b  = (_Float16*)alloc((size_t)4 * H_ * 1536 * 2);
  _Float16* Wdec0   = (_Float16*)alloc((size_t)4 * H_ * 1024 * 2);
  _Float16* Wdec1   = (_Float16*)alloc((size_t)4 * H_ * 1024 * 2);
  _Float16* Wmu16   = (_Float16*)alloc((size_t)L_ * 1024 * 2);
  _Float16* Wlv16   = (_Float16*)alloc((size_t)L_ * 1024 * 2);
  _Float16* Wdin16  = (_Float16*)alloc((size_t)H_ * L_ * 2);
  _Float16* Wout16  = (_Float16*)alloc((size_t)V_ * H_ * 2);
  _Float16* hzero   = (_Float16*)alloc((size_t)B_ * H_ * 2);
  _Float16* hp[2]   = {(_Float16*)alloc((size_t)B_ * H_ * 2),
                       (_Float16*)alloc((size_t)B_ * H_ * 2)};
  _Float16* hpd0[2] = {(_Float16*)alloc((size_t)B_ * H_ * 2),
                       (_Float16*)alloc((size_t)B_ * H_ * 2)};
  _Float16* hpd1[2] = {(_Float16*)alloc((size_t)B_ * H_ * 2),
                       (_Float16*)alloc((size_t)B_ * H_ * 2)};
  float*    c0      = (float*)alloc((size_t)B_ * H_ * 4);
  float*    cd0     = (float*)alloc((size_t)B_ * H_ * 4);
  float*    cd1     = (float*)alloc((size_t)B_ * H_ * 4);
  _Float16* hcat    = (_Float16*)alloc((size_t)B_ * 2 * H_ * 2);
  _Float16* mu16    = (_Float16*)alloc((size_t)B_ * L_ * 2);
  _Float16* h0dec   = (_Float16*)alloc((size_t)B_ * H_ * 2);
  _Float16* e16     = (_Float16*)alloc((size_t)B_ * H_ * 2);
  int*      tok     = (int*)alloc((size_t)B_ * 4);

  auto cvt = [&](_Float16* dst, int dpitch, int coloff, const float* src,
                 int rows, int cols) {
    int n = rows * cols;
    cvt2d_k<<<(n + 255) / 256, 256, 0, stream>>>(dst + coloff, dpitch, src, rows, cols);
  };

  // --- weight conversions + gate-weight concat [Wih | Whh] (f32 -> f16) ---
  cvt(emb16, H_, 0, emb, V_, H_);
  cvt(Wenc0f, 1024, 0,   enc_wih_l0,                       4 * H_, H_);
  cvt(Wenc0f, 1024, 512, enc_whh_l0,                       4 * H_, H_);
  cvt(Wenc0b, 1024, 0,   enc_wih_l0 + (size_t)4 * H_ * H_, 4 * H_, H_);
  cvt(Wenc0b, 1024, 512, enc_whh_l0 + (size_t)4 * H_ * H_, 4 * H_, H_);
  cvt(Wenc1f, 1536, 0,    enc_wih_l1,                            4 * H_, 2 * H_);
  cvt(Wenc1f, 1536, 1024, enc_whh_l1,                            4 * H_, H_);
  cvt(Wenc1b, 1536, 0,    enc_wih_l1 + (size_t)4 * H_ * 2 * H_,  4 * H_, 2 * H_);
  cvt(Wenc1b, 1536, 1024, enc_whh_l1 + (size_t)4 * H_ * H_,      4 * H_, H_);
  cvt(Wdec0, 1024, 0,   dec_wih,                       4 * H_, H_);
  cvt(Wdec0, 1024, 512, dec_whh,                       4 * H_, H_);
  cvt(Wdec1, 1024, 0,   dec_wih + (size_t)4 * H_ * H_, 4 * H_, H_);
  cvt(Wdec1, 1024, 512, dec_whh + (size_t)4 * H_ * H_, 4 * H_, H_);
  cvt(Wmu16,  1024, 0, fc_mu_w,   L_, 2 * H_);
  cvt(Wlv16,  1024, 0, fc_lv_w,   L_, 2 * H_);
  cvt(Wdin16, L_,   0, dec_in_w,  H_, L_);
  cvt(Wout16, H_,   0, dec_out_w, V_, H_);

  // --- embedding gather for encoder ---
  {
    int n = T_ * B_ * H_;
    embed_seq_k<<<(n + 255) / 256, 256, 0, stream>>>(seq16, x, emb16);
  }
  hipMemsetAsync(hzero, 0, (size_t)B_ * H_ * 2, stream);

  const dim3 cg(B_ / 32, H_ / 32), cb(128);

  // --- encoder layer 0, forward ---
  hipMemsetAsync(c0, 0, (size_t)B_ * H_ * 4, stream);
  for (int t = 0; t < T_; ++t) {
    lstm_cell_wmma<<<cg, cb, 0, stream>>>(
        seq16 + (size_t)t * B_ * H_, H_, H_,
        (t == 0) ? hzero : hp[(t - 1) & 1], Wenc0f, 1024, enc_b_l0,
        c0, hp[t & 1], seq1 + (size_t)t * B_ * 2 * H_, 2 * H_);
  }
  // --- encoder layer 0, backward ---
  hipMemsetAsync(c0, 0, (size_t)B_ * H_ * 4, stream);
  for (int s = 0; s < T_; ++s) {
    int t = T_ - 1 - s;
    lstm_cell_wmma<<<cg, cb, 0, stream>>>(
        seq16 + (size_t)t * B_ * H_, H_, H_,
        (s == 0) ? hzero : hp[(s - 1) & 1], Wenc0b, 1024, enc_b_l0 + 4 * H_,
        c0, hp[s & 1], seq1 + (size_t)t * B_ * 2 * H_ + H_, 2 * H_);
  }
  // --- encoder layer 1, forward (keep only final hidden -> hcat[:, :H]) ---
  hipMemsetAsync(c0, 0, (size_t)B_ * H_ * 4, stream);
  for (int t = 0; t < T_; ++t) {
    lstm_cell_wmma<<<cg, cb, 0, stream>>>(
        seq1 + (size_t)t * B_ * 2 * H_, 2 * H_, 2 * H_,
        (t == 0) ? hzero : hp[(t - 1) & 1], Wenc1f, 1536, enc_b_l1,
        c0, hp[t & 1], (t == T_ - 1) ? hcat : (_Float16*)nullptr, 2 * H_);
  }
  // --- encoder layer 1, backward (final hidden -> hcat[:, H:]) ---
  hipMemsetAsync(c0, 0, (size_t)B_ * H_ * 4, stream);
  for (int s = 0; s < T_; ++s) {
    int t = T_ - 1 - s;
    lstm_cell_wmma<<<cg, cb, 0, stream>>>(
        seq1 + (size_t)t * B_ * 2 * H_, 2 * H_, 2 * H_,
        (s == 0) ? hzero : hp[(s - 1) & 1], Wenc1b, 1536, enc_b_l1 + 4 * H_,
        c0, hp[s & 1], (s == T_ - 1) ? (hcat + H_) : (_Float16*)nullptr, 2 * H_);
  }

  // --- latent heads: mu (also f16 for decoder), logvar, decoder init ---
  float* mu_out = out + (size_t)B_ * T_ * V_;
  float* lv_out = mu_out + (size_t)B_ * L_;
  gemm_wmma<<<dim3(B_ / 32, L_ / 32), cb, 0, stream>>>(
      hcat, 2 * H_, Wmu16, 1024, 2 * H_, fc_mu_b, mu_out, L_, mu16, L_);
  gemm_wmma<<<dim3(B_ / 32, L_ / 32), cb, 0, stream>>>(
      hcat, 2 * H_, Wlv16, 1024, 2 * H_, fc_lv_b, lv_out, L_, nullptr, 0);
  gemm_wmma<<<dim3(B_ / 32, H_ / 32), cb, 0, stream>>>(
      mu16, L_, Wdin16, L_, L_, dec_in_b, nullptr, 0, h0dec, H_);

  // --- autoregressive decoder with argmax feedback ---
  hipMemsetAsync(cd0, 0, (size_t)B_ * H_ * 4, stream);
  hipMemsetAsync(cd1, 0, (size_t)B_ * H_ * 4, stream);
  fill_int_k<<<1, 256, 0, stream>>>(tok, 1, B_);
  for (int t = 0; t < T_; ++t) {
    embed_tok_k<<<(B_ * H_ + 255) / 256, 256, 0, stream>>>(e16, tok, emb16);
    lstm_cell_wmma<<<cg, cb, 0, stream>>>(
        e16, H_, H_,
        (t == 0) ? h0dec : hpd0[(t - 1) & 1], Wdec0, 1024, dec_b,
        cd0, hpd0[t & 1], nullptr, 0);
    lstm_cell_wmma<<<cg, cb, 0, stream>>>(
        hpd0[t & 1], H_, H_,
        (t == 0) ? h0dec : hpd1[(t - 1) & 1], Wdec1, 1024, dec_b + 4 * H_,
        cd1, hpd1[t & 1], nullptr, 0);
    // logits -> d_out[b, t, v] (row pitch T*V), then argmax feedback
    gemm_wmma<<<dim3(B_ / 32, V_ / 32), cb, 0, stream>>>(
        hpd1[t & 1], H_, Wout16, H_, H_, dec_out_b,
        out + (size_t)t * V_, T_ * V_, nullptr, 0);
    argmax_k<<<B_, 128, 0, stream>>>(tok, out + (size_t)t * V_, T_ * V_);
  }
}
